// Encoder_627065225609
// MI455X (gfx1250) — matrix operand order repaired and checked by reading, not compiled, not run on hardware
//
#include <hip/hip_runtime.h>

#define B_   4
#define N_   20480
#define C_   128
#define H_   8
#define HD_  16
#define L_   4
#define NW_  320
#define WS_  64
#define SCALE_ 0.25f   // HD^-0.5 = 16^-0.5

typedef __attribute__((ext_vector_type(16))) __bf16 v16bf;
typedef __attribute__((ext_vector_type(8)))  float  v8f;

union BFrag { v16bf v; uint4 q[2]; };

__device__ __forceinline__ unsigned short f2bf(float f) {
  unsigned int u = __float_as_uint(f);
  u += 0x7FFFu + ((u >> 16) & 1u);        // round-to-nearest-even
  return (unsigned short)(u >> 16);
}

// gelu(x) = 0.5x(1+tanh(u)) = x * t/(t+1),  t = exp(2u),
// u = 0.7978845608*(x + 0.044715 x^3)   (tanh-approx GELU, exact rewrite)
__device__ __forceinline__ float gelu_tanh(float x) {
  float t = __expf(1.5957691216057308f * (x + 0.044715f * x * x * x));
  return x * t / (t + 1.0f);
}

// ---------------------------------------------------------------------------
// Fragment loaders. All per-lane data is contiguous -> b128 vector loads.
//
// A fragment (16x32, row-major src[m][k]): lane holds M=lane&15;
//   k = 16*hf + j (contiguous)
__device__ __forceinline__ void load_a32(BFrag& f, const unsigned short* src,
                                         int stride, int row0, int k0) {
  const int lane = threadIdx.x & 31, hf = lane >> 4;
  const unsigned short* p = src + (size_t)(row0 + (lane & 15)) * stride + k0 + 16 * hf;
  f.q[0] = *(const uint4*)p;
  f.q[1] = *(const uint4*)(p + 8);
}

// A fragment with only K=0..15 valid (padded to 32 with zeros).
__device__ __forceinline__ void load_a16(BFrag& f, const unsigned short* src,
                                         int stride, int row0, int k0) {
  const int lane = threadIdx.x & 31, hf = lane >> 4;
  const unsigned short* p = src + (size_t)(row0 + (lane & 15)) * stride + k0;
  f.q[0] = hf ? make_uint4(0u, 0u, 0u, 0u) : *(const uint4*)p;
  f.q[1] = hf ? make_uint4(0u, 0u, 0u, 0u) : *(const uint4*)(p + 8);
}

// B fragment (32x16) from a TRANSPOSED source srcT[n][k] with row stride Ks.
//   lane holds N=lane&15; k = k0 + 16*hf + j  -> 16 contiguous elements.
__device__ __forceinline__ void load_bT(BFrag& f, const unsigned short* srcT,
                                        int Ks, int k0, int col0) {
  const int lane = threadIdx.x & 31, hf = lane >> 4;
  const unsigned short* p = srcT + (size_t)(col0 + (lane & 15)) * Ks + k0 + 16 * hf;
  f.q[0] = *(const uint4*)p;
  f.q[1] = *(const uint4*)(p + 8);
}

// B = K^T for attention scores: element (d, n) = qkvs[tok0+n][cbase+d],
// d valid 0..15 (K padded 16->32): low half loads a contiguous row slice,
// high half (d=16..31) is all zero.
__device__ __forceinline__ void load_b_scores(BFrag& f, const unsigned short* qkvs,
                                              int tok0, int cbase) {
  const int lane = threadIdx.x & 31, hf = lane >> 4;
  if (hf == 0) {
    const unsigned short* p = qkvs + (size_t)(tok0 + (lane & 15)) * 384 + cbase;
    f.q[0] = *(const uint4*)p;
    f.q[1] = *(const uint4*)(p + 8);
  } else {
    f.q[0] = make_uint4(0u, 0u, 0u, 0u);
    f.q[1] = make_uint4(0u, 0u, 0u, 0u);
  }
}

__device__ __forceinline__ v8f wmma_bf16(const BFrag& a, const BFrag& b, v8f c) {
  return __builtin_amdgcn_wmma_f32_16x16x32_bf16(false, a.v, false, b.v,
                                                 (short)0, c, false, false);
}

// ---------------------------------------------------------------------------
// Deterministic per-window argsort: 1 wave per window scans all ids in order.
__global__ void build_sort(const int* __restrict__ wid, int* __restrict__ sidx) {
  const int w = blockIdx.x;
  const int lane = threadIdx.x;
  int cnt = 0;
  for (int base = 0; base < N_; base += 32) {
    int id = wid[base + lane];
    unsigned int m32 = (unsigned int)__ballot(id == w);
    if (id == w) {
      int pos = cnt + __popc(m32 & ((1u << lane) - 1u));
      sidx[w * WS_ + pos] = base + lane;
    }
    cnt += __popc(m32);
  }
}

// fp32 [L][K][Nn] -> bf16 TRANSPOSED [L][Nn][K]
__global__ void convert_bf16_t(const float* __restrict__ src,
                               unsigned short* __restrict__ dst, int K, int Nn) {
  int i = blockIdx.x * 256 + threadIdx.x;
  int per = K * Nn;
  if (i >= L_ * per) return;
  int l = i / per, rem = i - l * per;
  int k = rem / Nn, nn = rem - k * Nn;
  dst[(size_t)l * per + (size_t)nn * K + k] = f2bf(src[i]);
}

// Shared LayerNorm body: 4 lanes per row, packed b128 bf16 output.
__device__ __forceinline__ void layernorm_to_bf16(
    const float* __restrict__ xrow_base, unsigned short* __restrict__ out,
    const float* __restrict__ g, const float* __restrict__ be, int tid) {
  int rr = tid >> 2, qq = tid & 3;
  const float* xr = xrow_base + rr * C_;
  float s = 0.f, s2 = 0.f;
#pragma unroll
  for (int c = 0; c < 32; ++c) { float v = xr[qq * 32 + c]; s += v; s2 += v * v; }
  s  += __shfl_xor(s, 1);  s  += __shfl_xor(s, 2);
  s2 += __shfl_xor(s2, 1); s2 += __shfl_xor(s2, 2);
  float mu  = s * (1.f / 128.f);
  float var = s2 * (1.f / 128.f) - mu * mu;
  float rs  = rsqrtf(var + 1e-5f);
  unsigned int pk[16];
#pragma unroll
  for (int c2 = 0; c2 < 16; ++c2) {
    int col = qq * 32 + c2 * 2;
    float v0 = (xr[col]     - mu) * rs * g[col]     + be[col];
    float v1 = (xr[col + 1] - mu) * rs * g[col + 1] + be[col + 1];
    pk[c2] = (unsigned int)f2bf(v0) | ((unsigned int)f2bf(v1) << 16);
  }
  uint4* dst = (uint4*)(out + rr * C_ + qq * 32);
#pragma unroll
  for (int j = 0; j < 4; ++j)
    dst[j] = make_uint4(pk[4 * j], pk[4 * j + 1], pk[4 * j + 2], pk[4 * j + 3]);
}

// ---------------------------------------------------------------------------
// Attention block: one workgroup per (batch, window). 8 waves, wave = head.
__global__ __launch_bounds__(256) void attn_kernel(
    float* __restrict__ xio,
    const float* __restrict__ g1, const float* __restrict__ be1,
    const unsigned short* __restrict__ WqkvT, const float* __restrict__ bqkv,
    const float* __restrict__ relb,
    const unsigned short* __restrict__ WprojT, const float* __restrict__ bproj,
    const int* __restrict__ sidx)
{
  extern __shared__ char smem[];
  float*          xw    = (float*)smem;                                 // 64x128 f32
  int*            sids  = (int*)(smem + 64 * 128 * 4);                  // 64 ints
  unsigned short* hbuf  = (unsigned short*)(smem + 64 * 128 * 4 + 256); // 64x128 bf16
  unsigned short* qkvs  = hbuf + 64 * 128;                              // 64x384 bf16
  unsigned short* pbuf  = qkvs + 64 * 384;                              // 8x16x64 bf16
  unsigned short* attnb = pbuf + 8 * 16 * 64;                           // 64x128 bf16
  unsigned short* vT    = attnb + 64 * 128;                             // 128x64 bf16 (V^T)

  const int tid = threadIdx.x;
  const int lane = tid & 31, wv = tid >> 5, hf = lane >> 4;
  const int bw = blockIdx.x, b = bw / NW_, w = bw % NW_;

  if (tid < WS_) sids[tid] = sidx[w * WS_ + tid];
  __syncthreads();

  // Gather the 64 window tokens; each thread owns one contiguous row span.
  {
    int row = tid >> 2, seg = tid & 3;
    const float4* src = (const float4*)(xio + ((size_t)b * N_ + sids[row]) * C_) + seg * 8;
    float4* dst = (float4*)(xw + row * C_) + seg * 8;
#pragma unroll
    for (int i = 0; i < 8; ++i) dst[i] = src[i];
  }
  __syncthreads();

  layernorm_to_bf16(xw, hbuf, g1, be1, tid);
  __syncthreads();

  // QKV GEMM: [64x128] @ [128x384]. Each wave owns 3 output column-tiles;
  // B fragments loaded once and reused across the 4 row tiles.
  for (int ci = 0; ci < 3; ++ci) {
    const int tn = wv + 8 * ci;
    v8f acc[4] = {{}, {}, {}, {}};
#pragma unroll
    for (int kk = 0; kk < 4; ++kk) {
      BFrag bb; load_bT(bb, WqkvT, C_, kk * 32, tn * 16);
#pragma unroll
      for (int tm = 0; tm < 4; ++tm) {
        BFrag a; load_a32(a, hbuf, C_, tm * 16, kk * 32);
        acc[tm] = wmma_bf16(a, bb, acc[tm]);
      }
    }
    const int col = tn * 16 + (lane & 15);
    const float bia = bqkv[col];
#pragma unroll
    for (int tm = 0; tm < 4; ++tm)
#pragma unroll
      for (int r = 0; r < 8; ++r) {
        int row = tm * 16 + r + 8 * hf;
        unsigned short val = f2bf(acc[tm][r] + bia);
        qkvs[row * 384 + col] = val;
        if (col >= 256) vT[(col - 256) * 64 + row] = val;   // V^T copy
      }
  }
  __syncthreads();

  // Attention: wave wv owns head h. q cols [h*16), k [128+h*16), v [256+h*16).
  const int h = wv;
  const float rb = relb[h];
  unsigned short* pw = pbuf + wv * 16 * 64;
  for (int qi = 0; qi < 4; ++qi) {
    BFrag aq;
    load_a16(aq, qkvs, 384, qi * 16, h * 16);     // K padded 16->32
    v8f sc0 = {}, sc1 = {}, sc2 = {}, sc3 = {};
    {
      BFrag bk;
      load_b_scores(bk, qkvs, 0,  128 + h * 16); sc0 = wmma_bf16(aq, bk, sc0);
      load_b_scores(bk, qkvs, 16, 128 + h * 16); sc1 = wmma_bf16(aq, bk, sc1);
      load_b_scores(bk, qkvs, 32, 128 + h * 16); sc2 = wmma_bf16(aq, bk, sc2);
      load_b_scores(bk, qkvs, 48, 128 + h * 16); sc3 = wmma_bf16(aq, bk, sc3);
    }
    // Row softmax across 4 column tiles + 16-lane groups.
#pragma unroll
    for (int r = 0; r < 8; ++r) {
      float v0 = sc0[r] * SCALE_ + rb, v1 = sc1[r] * SCALE_ + rb;
      float v2 = sc2[r] * SCALE_ + rb, v3 = sc3[r] * SCALE_ + rb;
      float mx = fmaxf(fmaxf(v0, v1), fmaxf(v2, v3));
      mx = fmaxf(mx, __shfl_xor(mx, 1)); mx = fmaxf(mx, __shfl_xor(mx, 2));
      mx = fmaxf(mx, __shfl_xor(mx, 4)); mx = fmaxf(mx, __shfl_xor(mx, 8));
      v0 = __expf(v0 - mx); v1 = __expf(v1 - mx);
      v2 = __expf(v2 - mx); v3 = __expf(v3 - mx);
      float sm = v0 + v1 + v2 + v3;
      sm += __shfl_xor(sm, 1); sm += __shfl_xor(sm, 2);
      sm += __shfl_xor(sm, 4); sm += __shfl_xor(sm, 8);
      float inv = 1.f / sm;
      int m = r + 8 * hf, n = lane & 15;
      pw[m * 64 +      n] = f2bf(v0 * inv);
      pw[m * 64 + 16 + n] = f2bf(v1 * inv);
      pw[m * 64 + 32 + n] = f2bf(v2 * inv);
      pw[m * 64 + 48 + n] = f2bf(v3 * inv);
    }
    __syncthreads();   // uniform; orders pbuf write -> read
    v8f o = {};
    {
      BFrag ap, bv;
      load_a32(ap, pw, 64, 0, 0);
      load_bT(bv, vT, 64, 0, h * 16);    // V^T: contiguous per lane
      o = wmma_bf16(ap, bv, o);
      load_a32(ap, pw, 64, 0, 32);
      load_bT(bv, vT, 64, 32, h * 16);
      o = wmma_bf16(ap, bv, o);
    }
#pragma unroll
    for (int r = 0; r < 8; ++r) {
      int row = qi * 16 + r + 8 * hf;
      attnb[row * C_ + h * 16 + (lane & 15)] = f2bf(o[r]);
    }
  }
  __syncthreads();

  // Output projection [64x128]@[128x128] + bias + residual, scatter to tokens.
  {
    const int tn = wv;                    // 8 column tiles, one per wave
    v8f acc[4] = {{}, {}, {}, {}};
#pragma unroll
    for (int kk = 0; kk < 4; ++kk) {
      BFrag bb; load_bT(bb, WprojT, C_, kk * 32, tn * 16);
#pragma unroll
      for (int tm = 0; tm < 4; ++tm) {
        BFrag a; load_a32(a, attnb, C_, tm * 16, kk * 32);
        acc[tm] = wmma_bf16(a, bb, acc[tm]);
      }
    }
    const int col = tn * 16 + (lane & 15);
    const float bia = bproj[col];
#pragma unroll
    for (int tm = 0; tm < 4; ++tm)
#pragma unroll
      for (int r = 0; r < 8; ++r) {
        int row = tm * 16 + r + 8 * hf;
        float val = acc[tm][r] + bia + xw[row * C_ + col];
        xio[((size_t)b * N_ + sids[row]) * C_ + col] = val;
      }
  }
}

// ---------------------------------------------------------------------------
// MLP block: one workgroup per 64-token tile; fused LN2 + GEMM1+GELU + GEMM2.
__global__ __launch_bounds__(256) void mlp_kernel(
    float* __restrict__ xio,
    const float* __restrict__ g2, const float* __restrict__ be2,
    const unsigned short* __restrict__ W1T, const float* __restrict__ b1,
    const unsigned short* __restrict__ W2T, const float* __restrict__ b2)
{
  extern __shared__ char smem[];
  float*          xt  = (float*)smem;                                // 64x128 f32
  unsigned short* h2  = (unsigned short*)(smem + 64 * 128 * 4);      // 64x128 bf16
  unsigned short* mid = h2 + 64 * 128;                               // 64x512 bf16

  const int tid = threadIdx.x, lane = tid & 31, wv = tid >> 5, hf = lane >> 4;
  const size_t t0 = (size_t)blockIdx.x * 64;

  {
    int row = tid >> 2, seg = tid & 3;
    const float4* src = (const float4*)(xio + (t0 + row) * C_) + seg * 8;
    float4* dst = (float4*)(xt + row * C_) + seg * 8;
#pragma unroll
    for (int i = 0; i < 8; ++i) dst[i] = src[i];
  }
  __syncthreads();

  layernorm_to_bf16(xt, h2, g2, be2, tid);
  __syncthreads();

  // mid = gelu(h2 @ W1 + b1): [64x128]@[128x512]. 4 column tiles per wave.
  for (int ci = 0; ci < 4; ++ci) {
    const int tn = wv + 8 * ci;
    v8f acc[4] = {{}, {}, {}, {}};
#pragma unroll
    for (int kk = 0; kk < 4; ++kk) {
      BFrag bb; load_bT(bb, W1T, C_, kk * 32, tn * 16);
#pragma unroll
      for (int tm = 0; tm < 4; ++tm) {
        BFrag a; load_a32(a, h2, C_, tm * 16, kk * 32);
        acc[tm] = wmma_bf16(a, bb, acc[tm]);
      }
    }
    const int col = tn * 16 + (lane & 15);
    const float bia = b1[col];
#pragma unroll
    for (int tm = 0; tm < 4; ++tm)
#pragma unroll
      for (int r = 0; r < 8; ++r) {
        int row = tm * 16 + r + 8 * hf;
        mid[row * 512 + col] = f2bf(gelu_tanh(acc[tm][r] + bia));
      }
  }
  __syncthreads();

  // out = mid @ W2 + b2 + residual: [64x512]@[512x128], in place.
  {
    const int tn = wv;                   // 8 column tiles, one per wave
    v8f acc[4] = {{}, {}, {}, {}};
#pragma unroll
    for (int kk = 0; kk < 16; ++kk) {
      BFrag bb; load_bT(bb, W2T, 4 * C_, kk * 32, tn * 16);
#pragma unroll
      for (int tm = 0; tm < 4; ++tm) {
        BFrag a; load_a32(a, mid, 512, tm * 16, kk * 32);
        acc[tm] = wmma_bf16(a, bb, acc[tm]);
      }
    }
    const int col = tn * 16 + (lane & 15);
    const float bia = b2[col];
#pragma unroll
    for (int tm = 0; tm < 4; ++tm)
#pragma unroll
      for (int r = 0; r < 8; ++r) {
        int row = tm * 16 + r + 8 * hf;
        xio[(t0 + row) * C_ + col] = acc[tm][r] + bia + xt[row * C_ + col];
      }
  }
}

// ---------------------------------------------------------------------------
extern "C" void kernel_launch(void* const* d_in, const int* in_sizes, int n_in,
                              void* d_out, int out_size, void* d_ws, size_t ws_size,
                              hipStream_t stream) {
  (void)in_sizes; (void)n_in; (void)out_size; (void)ws_size;
  const float* x     = (const float*)d_in[0];
  const float* g1    = (const float*)d_in[1];
  const float* be1   = (const float*)d_in[2];
  const float* Wqkv  = (const float*)d_in[3];
  const float* bqkv  = (const float*)d_in[4];
  const float* relb  = (const float*)d_in[5];
  const float* Wproj = (const float*)d_in[6];
  const float* bproj = (const float*)d_in[7];
  const float* g2    = (const float*)d_in[8];
  const float* be2   = (const float*)d_in[9];
  const float* W1    = (const float*)d_in[10];
  const float* b1    = (const float*)d_in[11];
  const float* W2    = (const float*)d_in[12];
  const float* b2    = (const float*)d_in[13];
  const int*   wid   = (const int*)d_in[14];
  float* out = (float*)d_out;

  // Workspace carve (transposed bf16 weights + sort index), ~1.7 MB.
  char* ws = (char*)d_ws;
  auto carve = [&](size_t bytes) {
    char* p = ws; ws += (bytes + 255) & ~(size_t)255; return p;
  };
  int*            sidx     = (int*)carve((size_t)N_ * 4);
  unsigned short* wqkv_bf  = (unsigned short*)carve((size_t)L_ * C_ * 3 * C_ * 2);
  unsigned short* wproj_bf = (unsigned short*)carve((size_t)L_ * C_ * C_ * 2);
  unsigned short* w1_bf    = (unsigned short*)carve((size_t)L_ * C_ * 4 * C_ * 2);
  unsigned short* w2_bf    = (unsigned short*)carve((size_t)L_ * 4 * C_ * C_ * 2);

  build_sort<<<NW_, 32, 0, stream>>>(wid, sidx);
  int n;
  n = L_ * C_ * 3 * C_;
  convert_bf16_t<<<(n + 255) / 256, 256, 0, stream>>>(Wqkv,  wqkv_bf,  C_, 3 * C_);
  n = L_ * C_ * C_;
  convert_bf16_t<<<(n + 255) / 256, 256, 0, stream>>>(Wproj, wproj_bf, C_, C_);
  n = L_ * C_ * 4 * C_;
  convert_bf16_t<<<(n + 255) / 256, 256, 0, stream>>>(W1,    w1_bf,    C_, 4 * C_);
  n = L_ * 4 * C_ * C_;
  convert_bf16_t<<<(n + 255) / 256, 256, 0, stream>>>(W2,    w2_bf,    4 * C_, C_);

  hipMemcpyAsync(out, x, (size_t)B_ * N_ * C_ * sizeof(float),
                 hipMemcpyDeviceToDevice, stream);

  constexpr size_t ATTN_SMEM = 64 * 128 * 4 + 256 +
      (size_t)(64 * 128 + 64 * 384 + 8 * 16 * 64 + 64 * 128 + 128 * 64) * 2; // 147712 B
  constexpr size_t MLP_SMEM  = 64 * 128 * 4 +
      (size_t)(64 * 128 + 64 * 512) * 2;                                     // 114688 B

  for (int l = 0; l < L_; ++l) {
    attn_kernel<<<B_ * NW_, 256, ATTN_SMEM, stream>>>(
        out, g1 + l * C_, be1 + l * C_,
        wqkv_bf + (size_t)l * C_ * 3 * C_, bqkv + l * 3 * C_,
        relb + l * H_,
        wproj_bf + (size_t)l * C_ * C_, bproj + l * C_, sidx);
    mlp_kernel<<<(B_ * N_) / 64, 256, MLP_SMEM, stream>>>(
        out, g2 + l * C_, be2 + l * C_,
        w1_bf + (size_t)l * C_ * 4 * C_, b1 + l * 4 * C_,
        w2_bf + (size_t)l * 4 * C_ * C_, b2 + l * C_);
  }
}
